// CrossAttention_43774306680837
// MI455X (gfx1250) — matrix operand order repaired and checked by reading, not compiled, not run on hardware
//
#include <hip/hip_runtime.h>

typedef _Float16 f16;
typedef __attribute__((ext_vector_type(16))) _Float16 v16h;
typedef __attribute__((ext_vector_type(8)))  float    v8f;
typedef __attribute__((ext_vector_type(4)))  unsigned int u32x4;
typedef __attribute__((ext_vector_type(8)))  int          i32x8;
typedef __attribute__((ext_vector_type(4)))  int          i32x4;

// ---------------------------------------------------------------------------
// WMMA fragment loaders (wave32); A and B both use contiguous k = half*16 + e, not the ISA table order.
// A 16x32 f16 tile, row-major with row stride ld:
//   lane: m = lane&15, half = lane>>4; VGPR v holds K = half*16 + v*2 (+1)
__device__ inline v16h load_a_frag(const f16* base, int ld) {
    int lane = threadIdx.x & 31;
    int half = lane >> 4, m = lane & 15;
    const f16* pr = base + (size_t)m * ld + half * 16;
    v16h a;
#pragma unroll
    for (int v = 0; v < 8; ++v) {
        int k0 = v * 2;
        a[2 * v]     = pr[k0];
        a[2 * v + 1] = pr[k0 + 1];
    }
    return a;
}

// B 32x16 (KxN) tile supplied TRANSPOSED as [N][K] rows (row stride ld):
//   lane: n = lane&15, K = (lane>>4)*16 .. +15 contiguous
__device__ inline v16h load_b_frag(const f16* baseT, int ld) {
    int lane = threadIdx.x & 31;
    int n = lane & 15, kb = (lane >> 4) * 16;
    const f16* pr = baseT + (size_t)n * ld + kb;
    v16h b;
#pragma unroll
    for (int e = 0; e < 16; ++e) b[e] = pr[e];
    return b;
}

__device__ inline v8f wmma_f16(v16h a, v16h b, v8f c) {
    return __builtin_amdgcn_wmma_f32_16x16x32_f16(false, a, false, b, (short)0, c,
                                                  false, false);
}

// ---------------------------------------------------------------------------
// Tensor Data Mover: 2D tile load (f16 elements) global -> LDS.
// D# packing per CDNA5 ISA:
//  g0: w0={count=1}, w1=lds_addr, w2=global_addr[31:0], w3=addr[56:32]|type=2<<30
//  g1: w0={data_size=1 (2B) at [17:16]}, w1[31:16]=tensor_dim0 lo,
//      w2=dim0 hi | dim1 lo<<16, w3=dim1 hi | tile_dim0<<16, w4=tile_dim1,
//      w5=tensor_dim0_stride lo32, w6/w7=0
// Toolchain uses the 6-arg builtin: (g0, g1, g2, g3, g4, cpol).
__device__ inline void tdm_load_2d_f16(unsigned lds_off, const f16* gaddr,
                                       unsigned tensor_k, unsigned tensor_m,
                                       unsigned tile_k, unsigned tile_m) {
    unsigned long long ga = (unsigned long long)(size_t)gaddr;
    u32x4 g0;
    g0[0] = 1u;                                       // count=1, user descriptor
    g0[1] = lds_off;                                  // lds_addr (bytes)
    g0[2] = (unsigned)(ga & 0xFFFFFFFFull);           // global_addr[31:0]
    g0[3] = (unsigned)((ga >> 32) & 0x1FFFFFFull) | (2u << 30);  // [56:32] | type=2
    i32x8 g1;
    g1[0] = (int)(1u << 16);                          // data_size=1 -> 2 bytes
    g1[1] = (int)((tensor_k & 0xFFFFu) << 16);        // tensor_dim0[15:0]
    g1[2] = (int)((tensor_k >> 16) | ((tensor_m & 0xFFFFu) << 16));
    g1[3] = (int)((tensor_m >> 16) | (tile_k << 16)); // tile_dim0 = tile_k
    g1[4] = (int)(tile_m & 0xFFFFu);                  // tile_dim1 = tile_m
    g1[5] = (int)tensor_k;                            // tensor_dim0_stride = K
    g1[6] = 0;
    g1[7] = 0;
    i32x4 z4 = {0, 0, 0, 0};
    i32x8 z8 = {0, 0, 0, 0, 0, 0, 0, 0};
    __builtin_amdgcn_tensor_load_to_lds(g0, g1, z4, z4, z8, 0);
}

// ---------------------------------------------------------------------------
// GEMM: C[MxN] = alpha * A[MxK](f16) @ Bt[NxK](f16, pre-transposed).
// Block tile 128x128, 256 threads = 8 waves (4 in M, 2 in N), wave tile 32x64,
// K chunks of 32. A tile staged via TDM (double buffered); B fragments read
// straight from global (lane-contiguous thanks to the pre-transpose).
template <int OUT_F16>
__global__ void gemm_wmma_t(const f16* __restrict__ A, const f16* __restrict__ Bt,
                            void* __restrict__ Cv, int M, int N, int K, float alpha) {
    __shared__ f16 As[2][128][32];
    int t = threadIdx.x;
    int lane = t & 31, wave = t >> 5;
    int wm = wave & 3, wn = wave >> 2;
    int m0 = blockIdx.y * 128, n0 = blockIdx.x * 128;
    v8f acc[2][4] = {};

    unsigned lds0 = (unsigned)(size_t)&As[0][0][0];
    unsigned lds1 = (unsigned)(size_t)&As[1][0][0];
    const int nIter = K / 32;

    if (wave == 0)
        tdm_load_2d_f16(lds0, A + (size_t)m0 * K, (unsigned)K, (unsigned)M, 32, 128);

    for (int it = 0; it < nIter; ++it) {
        __syncthreads();   // everyone done reading the buffer we are about to refill
        if (wave == 0) {
            if (it + 1 < nIter) {
                tdm_load_2d_f16((it & 1) ? lds0 : lds1,
                                A + (size_t)m0 * K + (it + 1) * 32,
                                (unsigned)K, (unsigned)M, 32, 128);
                __builtin_amdgcn_s_wait_tensorcnt((short)1);  // oldest (this chunk) done
            } else {
                __builtin_amdgcn_s_wait_tensorcnt((short)0);
            }
        }
        __syncthreads();   // publish staged A tile
        const f16* as = &As[it & 1][0][0];
        v16h a0 = load_a_frag(as + (size_t)(wm * 32) * 32, 32);
        v16h a1 = load_a_frag(as + (size_t)(wm * 32 + 16) * 32, 32);
        const f16* bbase = Bt + (size_t)(n0 + wn * 64) * K + it * 32;
#pragma unroll
        for (int nt = 0; nt < 4; ++nt) {
            v16h b = load_b_frag(bbase + (size_t)(nt * 16) * K, K);
            acc[0][nt] = wmma_f16(a0, b, acc[0][nt]);
            acc[1][nt] = wmma_f16(a1, b, acc[1][nt]);
        }
    }

    int nl = lane & 15, mh = (lane >> 4) * 8;
#pragma unroll
    for (int mi = 0; mi < 2; ++mi) {
#pragma unroll
        for (int v = 0; v < 8; ++v) {
            int gr = m0 + wm * 32 + mi * 16 + v + mh;
            size_t rb = (size_t)gr * N + n0 + wn * 64 + nl;
#pragma unroll
            for (int nt = 0; nt < 4; ++nt) {
                float xv = acc[mi][nt][v] * alpha;
                if (OUT_F16) ((f16*)Cv)[rb + nt * 16] = (f16)xv;
                else         ((float*)Cv)[rb + nt * 16] = xv;
            }
        }
    }
}

// ---------------------------------------------------------------------------
// Elementwise f32 -> f16 conversion (grid-stride).
__global__ void convert_f16(const float* __restrict__ in, f16* __restrict__ out, int n) {
    for (int i = blockIdx.x * 256 + threadIdx.x; i < n; i += gridDim.x * 256)
        out[i] = (f16)in[i];
}

// ---------------------------------------------------------------------------
// Transpose + convert: W[K][N] f32 -> Wt[N][K] f16. 32x32 tiles, block (32,8).
__global__ void transpose_f16(const float* __restrict__ W, f16* __restrict__ Wt,
                              int K, int N) {
    __shared__ float tile[32][33];
    int tx = threadIdx.x, ty = threadIdx.y;
    int n0 = blockIdx.x * 32, k0 = blockIdx.y * 32;
#pragma unroll
    for (int e = 0; e < 4; ++e)
        tile[ty + 8 * e][tx] = W[(size_t)(k0 + ty + 8 * e) * N + n0 + tx];
    __syncthreads();
#pragma unroll
    for (int e = 0; e < 4; ++e)
        Wt[(size_t)(n0 + ty + 8 * e) * K + k0 + tx] = (f16)tile[tx][ty + 8 * e];
}

// ---------------------------------------------------------------------------
// LayerNorm in place over rows of 4096 f16 values (f32 math). One block/row.
__global__ void ln_kernel(f16* __restrict__ h1, const float* __restrict__ g,
                          const float* __restrict__ bta) {
    __shared__ float red[256];
    int row = blockIdx.x, t = threadIdx.x;
    f16* p = h1 + (size_t)row * 4096;
    float s = 0.f, ss = 0.f;
#pragma unroll
    for (int e = 0; e < 16; ++e) {
        float x = (float)p[t + 256 * e];
        s += x; ss += x * x;
    }
    red[t] = s; __syncthreads();
    for (int o = 128; o > 0; o >>= 1) { if (t < o) red[t] += red[t + o]; __syncthreads(); }
    float mean = red[0] / 4096.f;
    __syncthreads();
    red[t] = ss; __syncthreads();
    for (int o = 128; o > 0; o >>= 1) { if (t < o) red[t] += red[t + o]; __syncthreads(); }
    float var  = red[0] / 4096.f - mean * mean;
    float rstd = rsqrtf(var + 1e-5f);
#pragma unroll
    for (int e = 0; e < 16; ++e) {
        int c = t + 256 * e;
        float x = (float)p[c];
        p[c] = (f16)((x - mean) * rstd * g[c] + bta[c]);
    }
}

// ---------------------------------------------------------------------------
// Raw attention logits: per (b,h) block. S[64x1024] = q_h[64x96] @ k_h[1024x96]^T.
// Scale already folded into q_h and k_h. A/B fragments read directly from global.
__global__ void scores_kernel(const f16* __restrict__ qh, const f16* __restrict__ kh,
                              f16* __restrict__ P) {
    int bh = blockIdx.x;            // b*8 + h
    int b = bh >> 3, h = bh & 7;
    int wave = threadIdx.x >> 5, lane = threadIdx.x & 31;
    const f16* qbase = qh + (size_t)(b * 64) * 768 + h * 96;
    const f16* kbase = kh + (size_t)(b * 1024 + wave * 128) * 768 + h * 96;
    f16* pout = P + (size_t)bh * 64 * 1024 + wave * 128;
#pragma unroll 1
    for (int mt = 0; mt < 4; ++mt) {
        v16h a0 = load_a_frag(qbase + (size_t)(mt * 16) * 768 +  0, 768);
        v16h a1 = load_a_frag(qbase + (size_t)(mt * 16) * 768 + 32, 768);
        v16h a2 = load_a_frag(qbase + (size_t)(mt * 16) * 768 + 64, 768);
#pragma unroll 1
        for (int nt = 0; nt < 8; ++nt) {
            v8f acc = {};
            acc = wmma_f16(a0, load_b_frag(kbase + (size_t)(nt * 16) * 768 +  0, 768), acc);
            acc = wmma_f16(a1, load_b_frag(kbase + (size_t)(nt * 16) * 768 + 32, 768), acc);
            acc = wmma_f16(a2, load_b_frag(kbase + (size_t)(nt * 16) * 768 + 64, 768), acc);
            int n = lane & 15, mh = (lane >> 4) * 8;
#pragma unroll
            for (int v = 0; v < 8; ++v)
                pout[(size_t)(mt * 16 + v + mh) * 1024 + nt * 16 + n] = (f16)acc[v];
        }
    }
}

// ---------------------------------------------------------------------------
// Row softmax in place over 1024 f16 values. One 128-thread block per row.
__global__ void softmax_kernel(f16* __restrict__ P) {
    __shared__ float red[128];
    int row = blockIdx.x, t = threadIdx.x;
    f16* p = P + (size_t)row * 1024;
    float xs[8], mx = -1e30f;
#pragma unroll
    for (int e = 0; e < 8; ++e) { xs[e] = (float)p[t + 128 * e]; mx = fmaxf(mx, xs[e]); }
    red[t] = mx; __syncthreads();
    for (int o = 64; o > 0; o >>= 1) { if (t < o) red[t] = fmaxf(red[t], red[t + o]); __syncthreads(); }
    mx = red[0]; __syncthreads();
    float s = 0.f;
#pragma unroll
    for (int e = 0; e < 8; ++e) { xs[e] = expf(xs[e] - mx); s += xs[e]; }
    red[t] = s; __syncthreads();
    for (int o = 64; o > 0; o >>= 1) { if (t < o) red[t] += red[t + o]; __syncthreads(); }
    float inv = 1.f / red[0];
#pragma unroll
    for (int e = 0; e < 8; ++e) p[t + 128 * e] = (f16)(xs[e] * inv);
}

// ---------------------------------------------------------------------------
// U[b,i,h,r] = sum_j P[b,h,i,j] * vmid[b,j,i,r]. Per (b,i) block, 128 threads.
// WMMA: M=16 (8 heads + zero pad), N=64 (r, one 16-tile per wave), K=1024 (j).
__global__ void pv_kernel(const f16* __restrict__ P, const f16* __restrict__ vmid,
                          float* __restrict__ U) {
    __shared__ f16 As[16][32];
    __shared__ f16 Vt[64][33];      // Vt[r][j] = vmid[b, j0+j, i, r]
    int bi = blockIdx.x;            // b*64 + i
    int b = bi >> 6, i = bi & 63;
    int t = threadIdx.x;
    int wave = t >> 5, lane = t & 31;
    // zero the padded A rows 8..15 once (never overwritten below)
    As[8 + (t >> 4)][(t & 15) * 2]     = (f16)0.f;
    As[8 + (t >> 4)][(t & 15) * 2 + 1] = (f16)0.f;

    const f16* pbase = P    + ((size_t)(b * 8) * 64 + i) * 1024;       // + h*64*1024 + j
    const f16* vbase = vmid + ((size_t)b * 1024 * 64 + i) * 64;        // + j*4096 + r
    v8f acc = {};
    for (int jc = 0; jc < 1024; jc += 32) {
        __syncthreads();
        if (t < 64) {               // stage P rows h=0..7, 32 j's
            int h = t >> 3, jj = (t & 7) * 4;
            const f16* src = pbase + (size_t)h * 64 * 1024 + jc + jj;
#pragma unroll
            for (int e = 0; e < 4; ++e) As[h][jj + e] = src[e];
        }
        {                           // stage vmid tile transposed
            int jj = t >> 2, r0 = (t & 3) * 16;
            const f16* src = vbase + (size_t)(jc + jj) * 4096 + r0;
#pragma unroll
            for (int e = 0; e < 16; ++e) Vt[r0 + e][jj] = src[e];
        }
        __syncthreads();
        v16h a  = load_a_frag(&As[0][0], 32);
        v16h bb = load_b_frag(&Vt[wave * 16][0], 33);
        acc = wmma_f16(a, bb, acc);
    }
    int n = lane & 15, mh = (lane >> 4) * 8;
    int r = wave * 16 + n;
#pragma unroll
    for (int v = 0; v < 8; ++v) {
        int m = v + mh;
        if (m < 8) U[((size_t)bi * 8 + m) * 64 + r] = acc[v];
    }
}

// ---------------------------------------------------------------------------
// out2[b,i,d] = sum_r U[b,i,d/96,r] * Wc[i,r,d]. Per (b,i) block, 256 threads.
__global__ void mix_kernel(const float* __restrict__ U, const float* __restrict__ Wc,
                           f16* __restrict__ out2) {
    __shared__ float Us[512];
    int bi = blockIdx.x;
    int i = bi & 63;
    int t = threadIdx.x;
    Us[t]       = U[(size_t)bi * 512 + t];
    Us[t + 256] = U[(size_t)bi * 512 + t + 256];
    __syncthreads();
    const float* wc = Wc + (size_t)i * 64 * 768;
#pragma unroll 1
    for (int d = t; d < 768; d += 256) {
        int h = d / 96;
        const float* uh = &Us[h * 64];
        float s = 0.f;
#pragma unroll
        for (int r = 0; r < 64; ++r) s += uh[r] * wc[(size_t)r * 768 + d];
        out2[(size_t)bi * 768 + d] = (f16)s;
    }
}

// ---------------------------------------------------------------------------
extern "C" void kernel_launch(void* const* d_in, const int* in_sizes, int n_in,
                              void* d_out, int out_size, void* d_ws, size_t ws_size,
                              hipStream_t stream) {
    const float* x       = (const float*)d_in[0];   // [4,64,768]
    const float* context = (const float*)d_in[1];   // [4,1024,768]
    const float* Wq      = (const float*)d_in[2];   // [768,768]
    const float* Wk      = (const float*)d_in[3];   // [768,768]
    const float* Wv1     = (const float*)d_in[4];   // [768,4096]
    const float* ln_g    = (const float*)d_in[5];   // [4096]
    const float* ln_b    = (const float*)d_in[6];   // [4096]
    const float* Wc      = (const float*)d_in[7];   // [64,64,768]
    const float* Wout    = (const float*)d_in[8];   // [768,768]
    float* out = (float*)d_out;                     // [4,64,768]

    char* ws = (char*)d_ws;
    f16* x_h    = (f16*)ws;   ws += (size_t)256 * 768 * 2;
    f16* ctx_h  = (f16*)ws;   ws += (size_t)4096 * 768 * 2;
    f16* Wqt    = (f16*)ws;   ws += (size_t)768 * 768 * 2;      // [N][K]
    f16* Wkt    = (f16*)ws;   ws += (size_t)768 * 768 * 2;
    f16* Wv1t   = (f16*)ws;   ws += (size_t)4096 * 768 * 2;
    f16* Woutt  = (f16*)ws;   ws += (size_t)768 * 768 * 2;
    f16* q_h    = (f16*)ws;   ws += (size_t)256 * 768 * 2;      // q*scale
    f16* k_h    = (f16*)ws;   ws += (size_t)4096 * 768 * 2;     // k*scale
    f16* h1     = (f16*)ws;   ws += (size_t)4096 * 4096 * 2;    // h1 -> vmid (in place)
    f16* P      = (f16*)ws;   ws += (size_t)32 * 64 * 1024 * 2; // logits -> probs
    float* U    = (float*)ws; ws += (size_t)256 * 8 * 64 * 4;   // [b,i,h,r]
    f16* out2   = (f16*)ws;   ws += (size_t)256 * 768 * 2;      // pre-Wout, f16

    const float scale = 0.3194722443f;  // 96^(-1/4), folded into both q and k

    // precision prep: f16 activations, f16 transposed weights
    convert_f16<<<768, 256, 0, stream>>>(x, x_h, 256 * 768);
    convert_f16<<<2048, 256, 0, stream>>>(context, ctx_h, 4096 * 768);
    transpose_f16<<<dim3(24, 24),  dim3(32, 8), 0, stream>>>(Wq,   Wqt,   768, 768);
    transpose_f16<<<dim3(24, 24),  dim3(32, 8), 0, stream>>>(Wk,   Wkt,   768, 768);
    transpose_f16<<<dim3(128, 24), dim3(32, 8), 0, stream>>>(Wv1,  Wv1t,  768, 4096);
    transpose_f16<<<dim3(24, 24),  dim3(32, 8), 0, stream>>>(Wout, Woutt, 768, 768);

    // projections + q-aware V mid (TDM-staged WMMA GEMMs)
    gemm_wmma_t<1><<<dim3(6, 2),   256, 0, stream>>>(x_h,   Wqt,  q_h, 256,  768,  768, scale);
    gemm_wmma_t<1><<<dim3(6, 32),  256, 0, stream>>>(ctx_h, Wkt,  k_h, 4096, 768,  768, scale);
    gemm_wmma_t<1><<<dim3(32, 32), 256, 0, stream>>>(ctx_h, Wv1t, h1,  4096, 4096, 768, 1.0f);
    ln_kernel<<<4096, 256, 0, stream>>>(h1, ln_g, ln_b);

    // attention (sum-swap: never materialize v)
    scores_kernel<<<32, 256, 0, stream>>>(q_h, k_h, P);
    softmax_kernel<<<2048, 128, 0, stream>>>(P);
    pv_kernel<<<256, 128, 0, stream>>>(P, h1, U);
    mix_kernel<<<256, 256, 0, stream>>>(U, Wc, out2);
    gemm_wmma_t<0><<<dim3(6, 2), 256, 0, stream>>>(out2, Woutt, out, 256, 768, 768, 1.0f);
}
